// MultiheadAttention_52003464020802
// MI455X (gfx1250) — matrix operand order repaired and checked by reading, not compiled, not run on hardware
//
#include <hip/hip_runtime.h>
#include <hip/hip_bf16.h>

typedef __bf16 bf16;
typedef __attribute__((ext_vector_type(16))) __bf16 v16bf;
typedef __attribute__((ext_vector_type(8)))  __bf16 v8bf;
typedef __attribute__((ext_vector_type(8)))  float  v8f;
typedef __attribute__((ext_vector_type(4)))  unsigned int v4u;
typedef __attribute__((ext_vector_type(8)))  int v8i;
typedef __attribute__((ext_vector_type(4)))  int v4i;

#define T_SEQ  2048
#define B_SZ   2
#define E_DIM  1024
#define H_NUM  16
#define D_HEAD 64
#define M_ROWS (T_SEQ * B_SZ)   // 4096
#define SCALING 0.125f          // D^-0.5

static __device__ __forceinline__ v16bf cat16(v8bf lo, v8bf hi) {
  return __builtin_shufflevector(lo, hi, 0,1,2,3,4,5,6,7,8,9,10,11,12,13,14,15);
}

// A-fragment (16x32 bf16, MxK) from LDS tile, row stride `stride` elems.
static __device__ __forceinline__ v16bf load_a_frag(const bf16* base, int stride, int lane) {
  const int row = lane & 15, half = lane >> 4;
  const bf16* p = base + row * stride;
  v8bf lo = *(const v8bf*)(p + half * 8);
  v8bf hi = *(const v8bf*)(p + 16 + half * 8);
  return cat16(lo, hi);
}

// B-fragment (32x16 bf16, KxN) from LDS stored n-major (base + n*stride + k).
static __device__ __forceinline__ v16bf load_b_frag(const bf16* base, int stride, int lane) {
  const int n = lane & 15, half = lane >> 4;
  const bf16* p = base + n * stride;
  v8bf lo = *(const v8bf*)(p + half * 8);
  v8bf hi = *(const v8bf*)(p + 16 + half * 8);
  return cat16(lo, hi);
}

// DPP16 row rotations: 16-lane all-reduce entirely in VALU (no LDS traffic).
// Wave32 has two 16-lane rows -> matches the per-half reduction domain.
template<int CTRL>
static __device__ __forceinline__ float dpp_rot(float x) {
  return __int_as_float(__builtin_amdgcn_update_dpp(
      0, __float_as_int(x), CTRL, 0xf, 0xf, true));
}
static __device__ __forceinline__ float row_max16(float x) {
  x = fmaxf(x, dpp_rot<0x121>(x));  // row_ror:1
  x = fmaxf(x, dpp_rot<0x122>(x));  // row_ror:2
  x = fmaxf(x, dpp_rot<0x124>(x));  // row_ror:4
  x = fmaxf(x, dpp_rot<0x128>(x));  // row_ror:8
  return x;
}
static __device__ __forceinline__ float row_sum16(float x) {
  x += dpp_rot<0x121>(x);
  x += dpp_rot<0x122>(x);
  x += dpp_rot<0x124>(x);
  x += dpp_rot<0x128>(x);
  return x;
}

// TDM: DMA a 2D bf16 tile (tile_d1 rows x tile_d0 elems, row stride stride0 elems)
// from global into LDS at lds_addr, padding 4 DWORDs every 32 DWORDs
// (=> LDS row stride of 72 bf16 for 64-elem rows). Tracked by TENSORcnt.
static __device__ __forceinline__ void tdm_load_2d_bf16(
    unsigned lds_addr, unsigned long long gaddr,
    unsigned tensor_d0, unsigned tensor_d1, unsigned long long stride0,
    unsigned tile_d0, unsigned tile_d1)
{
  v4u g0;
  g0.x = 1u;                                        // count=1 (valid user D#)
  g0.y = lds_addr;                                  // lds_addr [63:32]
  g0.z = (unsigned)(gaddr & 0xffffffffu);           // global_addr [95:64]
  g0.w = (unsigned)((gaddr >> 32) & 0x01ffffffu)    // global_addr [120:96]
       | (2u << 30);                                // type=2 ("image")
  v8i g1;
  g1[0] = (int)((1u << 16)      // data_size = 2 bytes
              | (1u << 20)      // pad_enable
              | (4u << 22)      // pad_interval: code 4 = 32 DWORDs
              | (3u << 25));    // pad_amount:   code 3 = 4 DWORDs (16B)
  g1[1] = (int)((tensor_d0 & 0xffffu) << 16);                      // tensor_dim0[15:0]
  g1[2] = (int)((tensor_d0 >> 16) | ((tensor_d1 & 0xffffu) << 16));
  g1[3] = (int)((tensor_d1 >> 16) | (tile_d0 << 16));              // tile_dim0
  g1[4] = (int)(tile_d1 & 0xffffu);                                // tile_dim1 (tile_dim2=0)
  g1[5] = (int)(stride0 & 0xffffffffu);                            // tensor_dim0_stride
  g1[6] = (int)((stride0 >> 32) & 0xffffu);
  g1[7] = 0;
  v4i g2 = {0, 0, 0, 0};
  v4i g3 = {0, 0, 0, 0};
#if defined(__clang_major__) && (__clang_major__ >= 23)
  v8i g4 = {0, 0, 0, 0, 0, 0, 0, 0};
  __builtin_amdgcn_tensor_load_to_lds(g0, g1, g2, g3, g4, 0);
#else
  __builtin_amdgcn_tensor_load_to_lds(g0, g1, g2, g3, 0);
#endif
}

// 64B global -> LDS via 4 async b128 transfers (per-lane addresses, ASYNCcnt).
static __device__ __forceinline__ void async_copy64(unsigned lds_addr,
                                                    unsigned long long gaddr)
{
  asm volatile(
      "global_load_async_to_lds_b128 %0, %1, off\n\t"
      "global_load_async_to_lds_b128 %0, %1, off offset:16\n\t"
      "global_load_async_to_lds_b128 %0, %1, off offset:32\n\t"
      "global_load_async_to_lds_b128 %0, %1, off offset:48"
      :: "v"(lds_addr), "v"(gaddr) : "memory");
}
static __device__ __forceinline__ void wait_async0() {
  asm volatile("s_wait_asynccnt 0" ::: "memory");
}

// C = (A @ W^T + bias) * scale.  A: [M,K] (f32 or bf16), W: [N,K] f32.
// LAYOUT: 0 = f32 [M,N]; 2 = bf16 split-head [B*H,T,D]; 3 = bf16 split-head-T [B*H,D,T]
// Double-buffered LDS; tile k+1 staged (global loads) under tile k's WMMAs.
template<bool A_BF16, int LAYOUT>
__global__ __launch_bounds__(128)
void gemm_bias_kernel(const void* __restrict__ Av, const float* __restrict__ W,
                      const float* __restrict__ bias, void* __restrict__ Cv,
                      int Kdim, int Ndim, float scale)
{
  __shared__ __align__(16) bf16 As[2][128][40];
  __shared__ __align__(16) bf16 Bs[2][128][40];

  const int tid  = threadIdx.x;
  const int lane = tid & 31, wave = tid >> 5;
  const int wr   = wave >> 1, wc = wave & 1;
  const int bm   = blockIdx.y * 128, bn = blockIdx.x * 128;
  const int NK   = Kdim >> 5;

  auto stage = [&](int buf, int kk) {
    if constexpr (A_BF16) {
      async_copy64((unsigned)(size_t)&As[buf][tid][0],
                   (unsigned long long)(size_t)((const bf16*)Av + (size_t)(bm + tid) * Kdim + kk));
    } else {
      const float4* src = (const float4*)((const float*)Av + (size_t)(bm + tid) * Kdim + kk);
      union { uint4 u4[4]; bf16 b[32]; } pk;
      #pragma unroll
      for (int i = 0; i < 8; ++i) {
        float4 f = src[i];
        pk.b[i*4+0] = (bf16)f.x; pk.b[i*4+1] = (bf16)f.y;
        pk.b[i*4+2] = (bf16)f.z; pk.b[i*4+3] = (bf16)f.w;
      }
      uint4* dst = (uint4*)&As[buf][tid][0];
      #pragma unroll
      for (int i = 0; i < 4; ++i) dst[i] = pk.u4[i];
    }
    {
      const float4* src = (const float4*)(W + (size_t)(bn + tid) * Kdim + kk);
      union { uint4 u4[4]; bf16 b[32]; } pk;
      #pragma unroll
      for (int i = 0; i < 8; ++i) {
        float4 f = src[i];
        pk.b[i*4+0] = (bf16)f.x; pk.b[i*4+1] = (bf16)f.y;
        pk.b[i*4+2] = (bf16)f.z; pk.b[i*4+3] = (bf16)f.w;
      }
      uint4* dst = (uint4*)&Bs[buf][tid][0];
      #pragma unroll
      for (int i = 0; i < 4; ++i) dst[i] = pk.u4[i];
    }
  };

  stage(0, 0);
  v8f acc[4][4] = {};

  for (int kt = 0; kt < NK; ++kt) {
    const int cur = kt & 1;
    if constexpr (A_BF16) wait_async0();
    __syncthreads();
    if (kt + 1 < NK) stage(cur ^ 1, (kt + 1) << 5);   // overlaps WMMAs below

    v16bf af[4], bf_[4];
    #pragma unroll
    for (int mi = 0; mi < 4; ++mi) af[mi]  = load_a_frag(&As[cur][wr*64 + mi*16][0], 40, lane);
    #pragma unroll
    for (int ni = 0; ni < 4; ++ni) bf_[ni] = load_b_frag(&Bs[cur][wc*64 + ni*16][0], 40, lane);
    #pragma unroll
    for (int mi = 0; mi < 4; ++mi)
      #pragma unroll
      for (int ni = 0; ni < 4; ++ni)
        acc[mi][ni] = __builtin_amdgcn_wmma_f32_16x16x32_bf16(
            false, af[mi], false, bf_[ni], (short)0, acc[mi][ni], false, false);
  }

  const int half = lane >> 4, lr = lane & 15;
  #pragma unroll
  for (int mi = 0; mi < 4; ++mi)
    #pragma unroll
    for (int ni = 0; ni < 4; ++ni)
      #pragma unroll
      for (int r = 0; r < 8; ++r) {
        int m = bm + wr*64 + mi*16 + half*8 + r;
        int n = bn + wc*64 + ni*16 + lr;
        float vv = (acc[mi][ni][r] + bias[n]) * scale;
        if constexpr (LAYOUT == 2) {
          int t = m / B_SZ, b = m % B_SZ;
          int h = n >> 6, d = n & 63;
          ((bf16*)Cv)[(((size_t)(b*H_NUM + h)) * T_SEQ + t) * D_HEAD + d] = (bf16)vv;
        } else if constexpr (LAYOUT == 3) {
          int t = m / B_SZ, b = m % B_SZ;
          int h = n >> 6, d = n & 63;
          ((bf16*)Cv)[(((size_t)(b*H_NUM + h)) * D_HEAD + d) * T_SEQ + t] = (bf16)vv;
        } else {
          ((float*)Cv)[(size_t)m * Ndim + n] = vv;
        }
      }
}

// Flash attention. q/k: bf16 [B*H, T, D]; v: bf16 [B*H, D, T] (pre-transposed).
// out: bf16 attn in [T, B, E] row order. bias: f32 [B*H, T, T]. kpm: [B, T].
// K/V tiles double-buffered via TDM: tile kt+1 DMA'd while tile kt computes.
__global__ __launch_bounds__(256)
void flash_attn_kernel(const bf16* __restrict__ q, const bf16* __restrict__ k,
                       const bf16* __restrict__ v, const float* __restrict__ bias,
                       const unsigned char* __restrict__ kpm, bf16* __restrict__ outp)
{
  __shared__ __align__(16) bf16 Qs[128][72];
  __shared__ __align__(16) bf16 Ks[2][64][72];
  __shared__ __align__(16) bf16 Vs[2][64][72];     // Vs[.][d][j]
  __shared__ __align__(16) bf16 Ps[8][16][72];     // per-wave P tile (wave-private)

  const int tid  = threadIdx.x;
  const int lane = tid & 31, wave = tid >> 5;
  const int half = lane >> 4, lr = lane & 15;
  const int q0   = blockIdx.x * 128;
  const int bh   = blockIdx.y;
  const int b_idx = bh / H_NUM, h_idx = bh % H_NUM;
  const int NT   = T_SEQ / 64;

  const bf16* kbase = k + (size_t)bh * T_SEQ * D_HEAD;
  const bf16* vbase = v + (size_t)bh * D_HEAD * T_SEQ;

  // kick off TDM for tile 0 before anything else
  if (wave == 0) {
    tdm_load_2d_bf16((unsigned)(size_t)&Ks[0][0][0],
                     (unsigned long long)(size_t)kbase,
                     64u, 64u, 64ull, 64u, 64u);
  } else if (wave == 1) {
    tdm_load_2d_bf16((unsigned)(size_t)&Vs[0][0][0],
                     (unsigned long long)(size_t)vbase,
                     64u, 64u, (unsigned long long)T_SEQ, 64u, 64u);
  }

  // stage Q block (async global->LDS, 64B per thread)
  {
    int row = tid >> 1, hf = tid & 1;
    async_copy64((unsigned)(size_t)&Qs[row][hf * 32],
                 (unsigned long long)(size_t)(q + ((size_t)bh * T_SEQ + q0 + row) * D_HEAD + hf * 32));
    wait_async0();
  }
  __syncthreads();

  const v16bf qa0 = load_a_frag(&Qs[wave*16][0],  72, lane);
  const v16bf qa1 = load_a_frag(&Qs[wave*16][32], 72, lane);

  float m_i[8], l_i[8];
  #pragma unroll
  for (int r = 0; r < 8; ++r) { m_i[r] = -1e30f; l_i[r] = 0.f; }
  v8f Oacc[4] = {};

  const float* bias_base =
      bias + ((size_t)bh * T_SEQ + q0 + wave*16 + half*8) * T_SEQ + lr;

  for (int kt = 0; kt < NT; ++kt) {
    const int cur = kt & 1;
    if (wave < 2) __builtin_amdgcn_s_wait_tensorcnt(0);   // tile kt landed
    __syncthreads();
    if (kt + 1 < NT) {                                    // prefetch tile kt+1
      if (wave == 0) {
        tdm_load_2d_bf16((unsigned)(size_t)&Ks[cur ^ 1][0][0],
                         (unsigned long long)(size_t)(kbase + (size_t)(kt + 1) * 64 * D_HEAD),
                         64u, 64u, 64ull, 64u, 64u);
      } else if (wave == 1) {
        tdm_load_2d_bf16((unsigned)(size_t)&Vs[cur ^ 1][0][0],
                         (unsigned long long)(size_t)(vbase + (size_t)(kt + 1) * 64),
                         64u, 64u, (unsigned long long)T_SEQ, 64u, 64u);
      }
    }

    // S = Q @ K^T  (q-scaling folded into projection)
    v8f S[4];
    const v8f z = {};
    #pragma unroll
    for (int ni = 0; ni < 4; ++ni) {
      v16bf b0 = load_b_frag(&Ks[cur][ni*16][0],  72, lane);
      v16bf b1 = load_b_frag(&Ks[cur][ni*16][32], 72, lane);
      v8f t0 = __builtin_amdgcn_wmma_f32_16x16x32_bf16(false, qa0, false, b0, (short)0, z,  false, false);
      S[ni]  = __builtin_amdgcn_wmma_f32_16x16x32_bf16(false, qa1, false, b1, (short)0, t0, false, false);
    }

    // + bias, key padding mask
    const float* bp = bias_base + kt * 64;
    #pragma unroll
    for (int ni = 0; ni < 4; ++ni) {
      bool msk = kpm[(size_t)b_idx * T_SEQ + kt*64 + ni*16 + lr] != 0;
      #pragma unroll
      for (int r = 0; r < 8; ++r) {
        float sv = S[ni][r] + bp[(size_t)r * T_SEQ + ni*16];
        S[ni][r] = msk ? -1e30f : sv;
      }
    }
    if (kt + 1 < NT) __builtin_prefetch(bp + 64, 0, 0);

    // online softmax (DPP16 row reductions, no LDS); P via wave-private LDS
    #pragma unroll
    for (int r = 0; r < 8; ++r) {
      float mx = row_max16(fmaxf(fmaxf(S[0][r], S[1][r]), fmaxf(S[2][r], S[3][r])));
      float mnew  = fmaxf(m_i[r], mx);
      float alpha = __expf(m_i[r] - mnew);
      float ps = 0.f;
      #pragma unroll
      for (int ni = 0; ni < 4; ++ni) {
        float p = __expf(S[ni][r] - mnew);
        S[ni][r] = p;
        ps += p;
      }
      ps = row_sum16(ps);
      l_i[r] = l_i[r] * alpha + ps;
      m_i[r] = mnew;
      #pragma unroll
      for (int nd = 0; nd < 4; ++nd) Oacc[nd][r] *= alpha;
      #pragma unroll
      for (int ni = 0; ni < 4; ++ni)
        Ps[wave][half*8 + r][ni*16 + lr] = (bf16)S[ni][r];
    }
    // no barrier needed: Ps is wave-private, LDS ops are in-order per wave

    // O += P @ V
    #pragma unroll
    for (int kc = 0; kc < 2; ++kc) {
      v16bf pa = load_a_frag(&Ps[wave][0][kc*32], 72, lane);
      #pragma unroll
      for (int nd = 0; nd < 4; ++nd) {
        v16bf vb = load_b_frag(&Vs[cur][nd*16][kc*32], 72, lane);
        Oacc[nd] = __builtin_amdgcn_wmma_f32_16x16x32_bf16(
            false, pa, false, vb, (short)0, Oacc[nd], false, false);
      }
    }
  }

  // normalize, merge heads into [T, B, E]
  #pragma unroll
  for (int r = 0; r < 8; ++r) {
    float inv = 1.0f / l_i[r];
    int t = q0 + wave*16 + half*8 + r;
    #pragma unroll
    for (int nd = 0; nd < 4; ++nd) {
      float vv = Oacc[nd][r] * inv;
      int col = h_idx * D_HEAD + nd*16 + lr;
      outp[((size_t)t * B_SZ + b_idx) * E_DIM + col] = (bf16)vv;
    }
  }
}

extern "C" void kernel_launch(void* const* d_in, const int* in_sizes, int n_in,
                              void* d_out, int out_size, void* d_ws, size_t ws_size,
                              hipStream_t stream)
{
  (void)in_sizes; (void)n_in; (void)out_size; (void)ws_size;
  const float* query     = (const float*)d_in[0];
  const float* key_      = (const float*)d_in[1];
  const float* value     = (const float*)d_in[2];
  const float* attn_bias = (const float*)d_in[3];
  const unsigned char* kpm = (const unsigned char*)d_in[4];
  const float* Wq = (const float*)d_in[5];
  const float* bq = (const float*)d_in[6];
  const float* Wk = (const float*)d_in[7];
  const float* bk = (const float*)d_in[8];
  const float* Wv = (const float*)d_in[9];
  const float* bv = (const float*)d_in[10];
  const float* Wo = (const float*)d_in[11];
  const float* bo = (const float*)d_in[12];
  float* out = (float*)d_out;

  bf16* q_ws = (bf16*)d_ws;                          // [B*H, T, D]
  bf16* k_ws = q_ws + (size_t)M_ROWS * E_DIM;        // [B*H, T, D]
  bf16* v_ws = k_ws + (size_t)M_ROWS * E_DIM;        // [B*H, D, T] (transposed)
  bf16* a_ws = v_ws + (size_t)M_ROWS * E_DIM;        // attn [T, B, E]

  dim3 ggrid(E_DIM / 128, M_ROWS / 128);
  gemm_bias_kernel<false, 2><<<ggrid, 128, 0, stream>>>(query, Wq, bq, q_ws, E_DIM, E_DIM, SCALING);
  gemm_bias_kernel<false, 2><<<ggrid, 128, 0, stream>>>(key_,  Wk, bk, k_ws, E_DIM, E_DIM, 1.0f);
  gemm_bias_kernel<false, 3><<<ggrid, 128, 0, stream>>>(value, Wv, bv, v_ws, E_DIM, E_DIM, 1.0f);

  dim3 fgrid(T_SEQ / 128, B_SZ * H_NUM);
  flash_attn_kernel<<<fgrid, 256, 0, stream>>>(q_ws, k_ws, v_ws, attn_bias, kpm, a_ws);

  gemm_bias_kernel<true, 0><<<ggrid, 128, 0, stream>>>(a_ws, Wo, bo, out, E_DIM, E_DIM, 1.0f);
}
